// AttentionGRUDecoder_4733053960787
// MI455X (gfx1250) — matrix operand order repaired and checked by reading, not compiled, not run on hardware
//
#include <hip/hip_runtime.h>
#include <hip/hip_bf16.h>
#include <math.h>

// ---------------- CDNA5 WMMA types ----------------
typedef __bf16  v16bf __attribute__((ext_vector_type(16)));
typedef float   v8f   __attribute__((ext_vector_type(8)));

#define DB 32     // batch
#define DT 128    // time steps
#define DH 1024   // hidden
#define DA 1024   // attention width (== encoder seq len)
#define DE 512    // encoder feature
#define DV 512    // vocab
#define DG 3072   // 3*H

#define NT 4      // N-tiles (16 cols each) per wave in the GEMM

// =====================================================================
// Weight packing: fp32 (K x N, row-major or transposed source) ->
// bf16 blocked WMMA-B layout. Block (kb,nt) = 512 bf16, stored lane-major:
//   elem index = lane*16 + j ;  n = nt*16 + (lane&15)
//   k = kb*32 + ((lane>=16)?16:0) + j        (B 32x16 layout)
// One wave (32 lanes) per block.
// =====================================================================
__global__ void pack_w_kernel(const float* __restrict__ src, int srcRowStride,
                              int transposed, int K, int N, __bf16* __restrict__ dst)
{
    const int lane = threadIdx.x;          // 0..31
    const int blk  = blockIdx.x;           // (K/32)*(N/16) blocks
    const int KB   = K >> 5;
    const int nt   = blk / KB;
    const int kb   = blk % KB;
    const int n    = nt * 16 + (lane & 15);
    const int kseg = kb * 32 + ((lane >= 16) ? 16 : 0);
    __bf16* out = dst + ((size_t)nt * KB + kb) * 512 + (size_t)lane * 16;
#pragma unroll
    for (int j = 0; j < 16; ++j) {
        const int k = kseg + j;
        const float v = transposed ? src[(size_t)n * srcRowStride + k]
                                   : src[(size_t)k * srcRowStride + n];
        out[j] = (__bf16)v;
    }
}

// =====================================================================
// GEMM: Out[M=32, N] = act( concat(A0[:,0:K0], A1[:,0:K1]) * Wblk + bias )
// A* are fp32 row-major with given row strides; converted to bf16 in-register.
// One wave per 16x(16*NT) output strip: the A fragment (16x32) is loaded and
// converted ONCE per K-chunk and fed to NT back-to-back
// v_wmma_f32_16x16x32_bf16 ops (fp32 accumulation).
// grid = (N/(16*NT), 2), block = 32 (full wave => EXEC all-ones for WMMA).
// =====================================================================
__global__ void gemm_bf16_kernel(const float* __restrict__ A0, int lda0, int K0,
                                 const float* __restrict__ A1, int lda1, int K1,
                                 const __bf16* __restrict__ Wblk,
                                 const float* __restrict__ bias,
                                 float* __restrict__ Out, int ldo, int relu)
{
    const int lane   = threadIdx.x;              // 0..31
    const int ntile0 = blockIdx.x * NT;          // first of NT N-tiles
    const int mtile  = blockIdx.y;               // 0..1  (M = 32)
    const int KB     = (K0 + K1) >> 5;
    const int m      = (lane & 15) + mtile * 16;
    const int koff   = (lane >= 16) ? 16 : 0;    // A 16x32 bf16 layout

    v8f c[NT];
#pragma unroll
    for (int t = 0; t < NT; ++t) c[t] = (v8f){};

    // block streams for the NT adjacent N-tiles (adjacent in blocked layout)
    const __bf16* wp = Wblk + (size_t)ntile0 * KB * 512 + (size_t)lane * 16;

    for (int kb = 0; kb < KB; ++kb) {
        const int kbase = kb << 5;
        const float* Ap;  int kloc, lda;
        if (kbase < K0) { Ap = A0; kloc = kbase;      lda = lda0; }
        else            { Ap = A1; kloc = kbase - K0; lda = lda1; }
        const float* arow = Ap + (size_t)m * lda + kloc + koff;

        // lane<16 holds K {0..15}; lane>=16 holds K {16..31}
        v16bf a;
#pragma unroll
        for (int j = 0; j < 8; ++j) {
            a[j]     = (__bf16)arow[j];        // contiguous 8 -> b128 x2
            a[j + 8] = (__bf16)arow[8 + j];    // contiguous 8 -> b128 x2
        }
#pragma unroll
        for (int t = 0; t < NT; ++t) {
            const v16bf b = *(const v16bf*)(wp + (size_t)t * KB * 512);
            c[t] = __builtin_amdgcn_wmma_f32_16x16x32_bf16(
                       false, a, false, b, (short)0, c[t], false, false);
        }
        wp += 512;
    }

    // C/D layout: n = lane&15 ; VGPR r -> M = r (+8 for lanes 16..31)
    const int mrow0 = mtile * 16 + ((lane >= 16) ? 8 : 0);
#pragma unroll
    for (int t = 0; t < NT; ++t) {
        const int n  = (ntile0 + t) * 16 + (lane & 15);
        const float bs = bias ? bias[n] : 0.0f;
#pragma unroll
        for (int r = 0; r < 8; ++r) {
            float v = c[t][r] + bs;
            if (relu) v = fmaxf(v, 0.0f);
            Out[(size_t)(mrow0 + r) * ldo + n] = v;
        }
    }
}

// =====================================================================
// Row softmax in-place, one block (256 threads) per batch row of length n.
// =====================================================================
__global__ void softmax_kernel(float* __restrict__ x, int n)
{
    __shared__ float sred[8];
    float* row = x + (size_t)blockIdx.x * n;
    const int tid = threadIdx.x;

    float mx = -INFINITY;
    for (int i = tid; i < n; i += 256) mx = fmaxf(mx, row[i]);
#pragma unroll
    for (int o = 16; o > 0; o >>= 1) mx = fmaxf(mx, __shfl_xor(mx, o));
    if ((tid & 31) == 0) sred[tid >> 5] = mx;
    __syncthreads();
    float rowmax = -INFINITY;
#pragma unroll
    for (int i = 0; i < 8; ++i) rowmax = fmaxf(rowmax, sred[i]);
    __syncthreads();

    float s = 0.0f;
    for (int i = tid; i < n; i += 256) {
        const float e = expf(row[i] - rowmax);
        row[i] = e;
        s += e;
    }
#pragma unroll
    for (int o = 16; o > 0; o >>= 1) s += __shfl_xor(s, o);
    if ((tid & 31) == 0) sred[tid >> 5] = s;
    __syncthreads();
    float tot = 0.0f;
#pragma unroll
    for (int i = 0; i < 8; ++i) tot += sred[i];
    const float inv = 1.0f / tot;
    for (int i = tid; i < n; i += 256) row[i] *= inv;
}

// =====================================================================
// ctx[b,e] = sum_l attn[b,l] * enc[b,l,e]   (B independent 1xA gemv)
// grid = B, block = 256 (each thread owns e and e+256); coalesced enc reads.
// =====================================================================
__global__ void ctx_kernel(const float* __restrict__ attn,
                           const float* __restrict__ enc,
                           float* __restrict__ ctx)
{
    const int b  = blockIdx.x;
    const int e0 = threadIdx.x;
    const float* ar = attn + (size_t)b * DA;
    const float* eb = enc  + (size_t)b * DA * DE;
    float acc0 = 0.0f, acc1 = 0.0f;
#pragma unroll 4
    for (int l = 0; l < DA; ++l) {
        const float s = ar[l];
        acc0 = fmaf(s, eb[(size_t)l * DE + e0],       acc0);
        acc1 = fmaf(s, eb[(size_t)l * DE + e0 + 256], acc1);
    }
    ctx[(size_t)b * DE + e0]       = acc0;
    ctx[(size_t)b * DE + e0 + 256] = acc1;
}

// =====================================================================
// GRU elementwise (torch gate order r,z,n), h updated in place.
// =====================================================================
__device__ __forceinline__ float sigmoidf(float x) { return 1.0f / (1.0f + expf(-x)); }

__global__ void gru_kernel(const float* __restrict__ gi,
                           const float* __restrict__ gh,
                           float* __restrict__ h)
{
    const int idx = blockIdx.x * 256 + threadIdx.x;     // < B*H
    const int b = idx >> 10, j = idx & 1023;
    const float* gib = gi + (size_t)b * DG;
    const float* ghb = gh + (size_t)b * DG;
    const float r = sigmoidf(gib[j]          + ghb[j]);
    const float z = sigmoidf(gib[DH + j]     + ghb[DH + j]);
    const float n = tanhf   (gib[2 * DH + j] + r * ghb[2 * DH + j]);
    const float hold = h[idx];
    h[idx] = (1.0f - z) * n + z * hold;
}

// init: h <- hidden[0], zdec <- 0
__global__ void init_kernel(const float* __restrict__ hidden,
                            float* __restrict__ h, float* __restrict__ zdec)
{
    const int idx = blockIdx.x * 256 + threadIdx.x;     // < B*H
    h[idx]    = hidden[idx];
    zdec[idx] = 0.0f;
}

// =====================================================================
// Host orchestration
// =====================================================================
extern "C" void kernel_launch(void* const* d_in, const int* in_sizes, int n_in,
                              void* d_out, int out_size, void* d_ws, size_t ws_size,
                              hipStream_t stream)
{
    (void)in_sizes; (void)n_in; (void)out_size; (void)ws_size;

    const float* enc    = (const float*)d_in[0];   // [B,A,E]
    const float* hidden = (const float*)d_in[1];   // [1,B,H]
    const float* target = (const float*)d_in[2];   // [B,T,H]
    const float* W_att  = (const float*)d_in[3];   // [H+A, A]
    const float* b_att  = (const float*)d_in[4];
    const float* W_comb = (const float*)d_in[5];   // [H+E, H]
    const float* b_comb = (const float*)d_in[6];
    const float* W_ih   = (const float*)d_in[7];   // [3H, H]  (used transposed)
    const float* b_ih   = (const float*)d_in[8];
    const float* W_hh   = (const float*)d_in[9];   // [3H, H]  (used transposed)
    const float* b_hh   = (const float*)d_in[10];
    const float* W_out  = (const float*)d_in[11];  // [H, V]
    const float* b_out  = (const float*)d_in[12];
    float* out = (float*)d_out;                    // [B,T,V] then [1,B,H]

    // ---- workspace carve-up (256B aligned) ----
    char* ws = (char*)d_ws;
    size_t off = 0;
    auto take = [&](size_t bytes) -> char* {
        char* p = ws + off;
        off = (off + bytes + 255) & ~(size_t)255;
        return p;
    };
    __bf16* wa_blk   = (__bf16*)take((size_t)(DH + DA) * DA * 2);
    __bf16* wc_blk   = (__bf16*)take((size_t)(DH + DE) * DH * 2);
    __bf16* wih_blk  = (__bf16*)take((size_t)DH * DG * 2);
    __bf16* whh_blk  = (__bf16*)take((size_t)DH * DG * 2);
    __bf16* wout_blk = (__bf16*)take((size_t)DH * DV * 2);
    float*  hbuf     = (float*)take((size_t)DB * DH * 4);
    float*  zdec     = (float*)take((size_t)DB * DH * 4);
    float*  logits   = (float*)take((size_t)DB * DA * 4);   // softmax in place
    float*  ctxb     = (float*)take((size_t)DB * DE * 4);
    float*  gbuf     = (float*)take((size_t)DB * DH * 4);
    float*  gibuf    = (float*)take((size_t)DB * DG * 4);
    float*  ghbuf    = (float*)take((size_t)DB * DG * 4);

    const dim3 wave(32);

    // ---- pack all weights to bf16 WMMA-B blocked layout (once per call) ----
    pack_w_kernel<<<dim3(((DH + DA) / 32) * (DA / 16)), wave, 0, stream>>>(
        W_att, DA, 0, DH + DA, DA, wa_blk);
    pack_w_kernel<<<dim3(((DH + DE) / 32) * (DH / 16)), wave, 0, stream>>>(
        W_comb, DH, 0, DH + DE, DH, wc_blk);
    pack_w_kernel<<<dim3((DH / 32) * (DG / 16)), wave, 0, stream>>>(
        W_ih, DH, 1, DH, DG, wih_blk);                       // transposed use
    pack_w_kernel<<<dim3((DH / 32) * (DG / 16)), wave, 0, stream>>>(
        W_hh, DH, 1, DH, DG, whh_blk);                       // transposed use
    pack_w_kernel<<<dim3((DH / 32) * (DV / 16)), wave, 0, stream>>>(
        W_out, DV, 0, DH, DV, wout_blk);

    init_kernel<<<dim3(DB * DH / 256), dim3(256), 0, stream>>>(hidden, hbuf, zdec);

    // ---- recurrence ----
    for (int t = 0; t < DT; ++t) {
        const float* dec  = (t == 0) ? zdec : (target + (size_t)(t - 1) * DH);
        const int    ldd  = (t == 0) ? DH : DT * DH;   // row stride over batch

        // attn logits = [dec, h] @ W_att + b_att        -> [B, A]
        gemm_bf16_kernel<<<dim3(DA / (16 * NT), 2), wave, 0, stream>>>(
            dec, ldd, DH, hbuf, DH, DH, wa_blk, b_att, logits, DA, 0);

        softmax_kernel<<<dim3(DB), dim3(256), 0, stream>>>(logits, DA);

        ctx_kernel<<<dim3(DB), dim3(256), 0, stream>>>(logits, enc, ctxb);

        // g = relu([dec, ctx] @ W_comb + b_comb)        -> [B, H]
        gemm_bf16_kernel<<<dim3(DH / (16 * NT), 2), wave, 0, stream>>>(
            dec, ldd, DH, ctxb, DE, DE, wc_blk, b_comb, gbuf, DH, 1);

        // gi = g @ W_ih^T + b_ih ; gh = h @ W_hh^T + b_hh   -> [B, 3H]
        gemm_bf16_kernel<<<dim3(DG / (16 * NT), 2), wave, 0, stream>>>(
            gbuf, DH, DH, nullptr, 0, 0, wih_blk, b_ih, gibuf, DG, 0);
        gemm_bf16_kernel<<<dim3(DG / (16 * NT), 2), wave, 0, stream>>>(
            hbuf, DH, DH, nullptr, 0, 0, whh_blk, b_hh, ghbuf, DG, 0);

        gru_kernel<<<dim3(DB * DH / 256), dim3(256), 0, stream>>>(gibuf, ghbuf, hbuf);

        // out[:, t, :] = h_new @ W_out + b_out          -> [B, V]
        gemm_bf16_kernel<<<dim3(DV / (16 * NT), 2), wave, 0, stream>>>(
            hbuf, DH, DH, nullptr, 0, 0, wout_blk, b_out,
            out + (size_t)t * DV, DT * DV, 0);
    }

    // append h_last [1,B,H]
    hipMemcpyAsync(out + (size_t)DB * DT * DV, hbuf, (size_t)DB * DH * sizeof(float),
                   hipMemcpyDeviceToDevice, stream);
}
